// GCN_36541581754948
// MI455X (gfx1250) — matrix operand order repaired and checked by reading, not compiled, not run on hardware
//
#include <hip/hip_runtime.h>

typedef __attribute__((ext_vector_type(4)))  _Float16 h4;
typedef __attribute__((ext_vector_type(8)))  _Float16 h8;
typedef __attribute__((ext_vector_type(16))) _Float16 v16h;
typedef __attribute__((ext_vector_type(8)))  float    v8f;

#define IN_F  128
#define HID_F 64
#define SA    136   // LDS A-tile row stride in halves (128 + 8 pad -> shifts banks by 4/row)

// ---------------- degree / normalization ----------------
__global__ void k_init_deg(float* __restrict__ deg, int n) {
  int i = blockIdx.x * blockDim.x + threadIdx.x;
  if (i < n) deg[i] = 1.0f;                      // self-loop contributes 1
}

__global__ void k_count_deg(const long long* __restrict__ ei, int E, float* __restrict__ deg) {
  int e = blockIdx.x * blockDim.x + threadIdx.x;
  if (e < E) atomicAdd(&deg[(int)ei[(size_t)E + e]], 1.0f);
}

__global__ void k_rsqrt(float* __restrict__ deg, int n) {
  int i = blockIdx.x * blockDim.x + threadIdx.x;
  if (i < n) deg[i] = rsqrtf(deg[i]);            // in place: deg -> dis
}

// ---------------- W1 -> f16 (once) ----------------
__global__ void k_cvt_w1(const float* __restrict__ W1, _Float16* __restrict__ w1h) {
  int i = blockIdx.x * blockDim.x + threadIdx.x;
  if (i < HID_F * IN_F) w1h[i] = (_Float16)W1[i];
}

// ---------------- layer-1 GEMM via WMMA ----------------
// block = 128 threads = 4 waves. Block computes rows [16b, 16b+16) x all 64 cols.
// Wave w computes the 16x16 tile at col_base = 16w. A tile shared via LDS (f16).
// Epilogue: hs = (x @ W1^T) * dis[row]  (pre-scaled messages), agg seeded = hs (self loop).
__global__ __launch_bounds__(128)
void k_gemm1(const float* __restrict__ x, const _Float16* __restrict__ w1h,
             const float* __restrict__ dis, float* __restrict__ hs,
             float* __restrict__ agg, int N) {
  __shared__ _Float16 tA[16 * SA];
  const int tid = threadIdx.x;
  const int row_base = blockIdx.x * 16;

  // cooperative load + f32->f16 convert of the 16x128 A tile (16 floats / thread)
  {
    const int r = tid >> 3;                      // tile row 0..15
    const int g = tid & 7;                       // 16-float group 0..7
    int row = row_base + r; if (row >= N) row = N - 1;
    const float4* s4 = (const float4*)(x + (size_t)row * IN_F + g * 16);
    h4* d4 = (h4*)(tA + r * SA + g * 16);
#pragma unroll
    for (int j = 0; j < 4; ++j) {
      float4 v = s4[j];
      h4 h; h[0] = (_Float16)v.x; h[1] = (_Float16)v.y;
            h[2] = (_Float16)v.z; h[3] = (_Float16)v.w;
      d4[j] = h;
    }
  }
  __syncthreads();

  const int lane = tid & 31;
  const int wvid = tid >> 5;
  const int col_base = wvid * 16;
  const int m  = lane & 15;                      // A row / B col / D col index
  const int hi = lane >> 4;                      // lane half
  const int koff_a = hi * 8;                     // A: K 0-7/16-23 vs 8-15/24-31
  const int koff_b = hi * 8;                     // B: K 0-7/16-23 vs 8-15/24-31

  v8f c = {};
#pragma unroll
  for (int kb = 0; kb < IN_F; kb += 32) {
    const _Float16* ap = tA + m * SA + kb + koff_a;
    h8 alo = *(const h8*)ap;                     // K = kb+koff_a .. +7
    h8 ahi = *(const h8*)(ap + 16);              // K = kb+16+koff_a .. +7
    v16h a;
#pragma unroll
    for (int e = 0; e < 8; ++e) { a[e] = alo[e]; a[8 + e] = ahi[e]; }

    // B[k][n] = W1[n][k]; lane n=m, element e<8 -> K = kb + koff_b + e, e>=8 -> K = kb + 16 + koff_b + (e-8)
    const _Float16* bp = w1h + (size_t)(col_base + m) * IN_F + kb + koff_b;
    h8 blo = *(const h8*)bp;
    h8 bhi = *(const h8*)(bp + 16);
    v16h b;
#pragma unroll
    for (int e = 0; e < 8; ++e) { b[e] = blo[e]; b[8 + e] = bhi[e]; }

    c = __builtin_amdgcn_wmma_f32_16x16x32_f16(false, a, false, b,
                                               (short)0, c, false, false);
  }

  // D layout: VGPR r -> M = r + 8*hi, N = lane&15
  const int mrow0 = row_base + hi * 8;           // lane covers rows mrow0 .. mrow0+7
  if (row_base + 16 <= N) {
    // fast path: whole tile in-bounds (always taken when N % 16 == 0)
    float4 d0 = *(const float4*)(dis + mrow0);
    float4 d1 = *(const float4*)(dis + mrow0 + 4);
    float dsc[8] = {d0.x, d0.y, d0.z, d0.w, d1.x, d1.y, d1.z, d1.w};
    float* ho = hs  + (size_t)mrow0 * HID_F + col_base + m;
    float* ao = agg + (size_t)mrow0 * HID_F + col_base + m;
#pragma unroll
    for (int r = 0; r < 8; ++r) {
      float v = c[r] * dsc[r];                   // pre-scale by dis[src]
      ho[r * HID_F] = v;
      ao[r * HID_F] = v;                         // seed accumulator with self-loop term
    }
  } else {
#pragma unroll
    for (int r = 0; r < 8; ++r) {
      int row = mrow0 + r;
      if (row < N) {
        float v = c[r] * dis[row];
        size_t idx = (size_t)row * HID_F + col_base + m;
        hs[idx]  = v;
        agg[idx] = v;
      }
    }
  }
}

// ---------------- layer-1 edge scatter ----------------
// 16 threads per edge, 4 features each: agg[dst] += hs[src]
__global__ void k_scatter1(const long long* __restrict__ ei, int E,
                           const float* __restrict__ hs, float* __restrict__ agg) {
  int t = blockIdx.x * blockDim.x + threadIdx.x;
  int e = t >> 4;
  int f = (t & 15) * 4;
  if (e < E) {
    int s = (int)ei[e];
    int d = (int)ei[(size_t)E + e];
    float4 v = *(const float4*)(hs + (size_t)s * HID_F + f);
    float* dst = agg + (size_t)d * HID_F + f;
    atomicAdd(dst + 0, v.x);
    atomicAdd(dst + 1, v.y);
    atomicAdd(dst + 2, v.z);
    atomicAdd(dst + 3, v.w);
  }
}

// ---------------- layer-1 epilogue: hr = relu(dis[dst]*agg + b1) ----------------
__global__ void k_finish1(const float* __restrict__ agg, const float* __restrict__ dis,
                          const float* __restrict__ b1, float* __restrict__ hr, int N) {
  int t = blockIdx.x * blockDim.x + threadIdx.x;
  int i = t >> 4;
  int f = (t & 15) * 4;
  if (i < N) {
    float d = dis[i];
    float4 a  = *(const float4*)(agg + (size_t)i * HID_F + f);
    float4 bb = *(const float4*)(b1 + f);
    float4 o;
    o.x = fmaxf(0.f, d * a.x + bb.x);
    o.y = fmaxf(0.f, d * a.y + bb.y);
    o.z = fmaxf(0.f, d * a.z + bb.z);
    o.w = fmaxf(0.f, d * a.w + bb.w);
    *(float4*)(hr + (size_t)i * HID_F + f) = o;
  }
}

// ---------------- layer 2: per-node dot, pre-scaled; seeds d_out ----------------
__global__ void k_dot2(const float* __restrict__ hr, const float* __restrict__ W2,
                       const float* __restrict__ dis, float* __restrict__ h2s,
                       float* __restrict__ outacc, int N) {
  int i = blockIdx.x * blockDim.x + threadIdx.x;
  if (i < N) {
    const float4* a = (const float4*)(hr + (size_t)i * HID_F);
    const float4* w = (const float4*)W2;
    float acc = 0.f;
#pragma unroll
    for (int j = 0; j < HID_F / 4; ++j) {
      float4 av = a[j], wv = w[j];
      acc += av.x * wv.x + av.y * wv.y + av.z * wv.z + av.w * wv.w;
    }
    float v = acc * dis[i];
    h2s[i]    = v;
    outacc[i] = v;                               // self-loop seed (also inits d_out)
  }
}

__global__ void k_scatter2(const long long* __restrict__ ei, int E,
                           const float* __restrict__ h2s, float* __restrict__ outacc) {
  int e = blockIdx.x * blockDim.x + threadIdx.x;
  if (e < E) atomicAdd(&outacc[(int)ei[(size_t)E + e]], h2s[(int)ei[e]]);
}

__global__ void k_finish2(float* __restrict__ outacc, const float* __restrict__ dis,
                          const float* __restrict__ b2, int N) {
  int i = blockIdx.x * blockDim.x + threadIdx.x;
  if (i < N) outacc[i] = dis[i] * outacc[i] + b2[0];
}

// ---------------- launch ----------------
extern "C" void kernel_launch(void* const* d_in, const int* in_sizes, int n_in,
                              void* d_out, int out_size, void* d_ws, size_t ws_size,
                              hipStream_t stream) {
  const float*     x  = (const float*)d_in[0];
  const long long* ei = (const long long*)d_in[1];
  const float*     W1 = (const float*)d_in[2];
  const float*     b1 = (const float*)d_in[3];
  const float*     W2 = (const float*)d_in[4];
  const float*     b2 = (const float*)d_in[5];
  float*           out = (float*)d_out;

  const int N = in_sizes[0] / IN_F;
  const int E = in_sizes[1] / 2;

  // workspace partition (256B-aligned regions)
  char* p = (char*)d_ws;
  auto take = [&](size_t bytes) { char* r = p; p += (bytes + 255) & ~(size_t)255; return r; };
  float*    dis = (float*)take(sizeof(float) * (size_t)N);            // deg -> dis (in place)
  float*    hs  = (float*)take(sizeof(float) * (size_t)N * HID_F);    // pre-scaled h, then hr
  float*    agg = (float*)take(sizeof(float) * (size_t)N * HID_F);    // layer-1 accumulator
  float*    h2s = (float*)take(sizeof(float) * (size_t)N);            // pre-scaled layer-2 scalar
  _Float16* w1h = (_Float16*)take(sizeof(_Float16) * HID_F * IN_F);   // W1 in f16

  const int T = 256;
  k_init_deg <<<(N + T - 1) / T, T, 0, stream>>>(dis, N);
  k_count_deg<<<(E + T - 1) / T, T, 0, stream>>>(ei, E, dis);
  k_rsqrt    <<<(N + T - 1) / T, T, 0, stream>>>(dis, N);
  k_cvt_w1   <<<(HID_F * IN_F + T - 1) / T, T, 0, stream>>>(W1, w1h);

  k_gemm1    <<<(N + 15) / 16, 128, 0, stream>>>(x, w1h, dis, hs, agg, N);
  k_scatter1 <<<(E * 16 + T - 1) / T, T, 0, stream>>>(ei, E, hs, agg);
  k_finish1  <<<(N * 16 + T - 1) / T, T, 0, stream>>>(agg, dis, b1, hs, N);

  k_dot2     <<<(N + T - 1) / T, T, 0, stream>>>(hs, W2, dis, h2s, out, N);
  k_scatter2 <<<(E + T - 1) / T, T, 0, stream>>>(ei, E, h2s, out);
  k_finish2  <<<(N + T - 1) / T, T, 0, stream>>>(out, dis, b2, N);
}
